// MultiHeadedAttention_7627861917917
// MI455X (gfx1250) — compile-verified, matrix operand order checked by reading, not run on hardware
//
#include <hip/hip_runtime.h>

// ---------- types ----------
typedef __attribute__((ext_vector_type(16))) __bf16        v16bf;
typedef __attribute__((ext_vector_type(8)))  float         v8f;
typedef __attribute__((ext_vector_type(4)))  unsigned int  u32x4;
typedef __attribute__((ext_vector_type(4)))  float         f32x4;

// shapes (fixed by reference)
#define BB 2
#define TT 2048
#define DD 1024
#define HH 16
#define DK 64

// f32 -> bf16 via native fptrunc (RNE); backend picks v_cvt hardware op
__device__ __forceinline__ unsigned short f2bf(float f) {
    __bf16 h = (__bf16)f;
    return __builtin_bit_cast(unsigned short, h);
}

// packed f32x2 -> bf16x2 in one dword
__device__ __forceinline__ unsigned f2bf2(float a, float b) {
#if __has_builtin(__builtin_amdgcn_cvt_pk_bf16_f32)
    typedef __attribute__((ext_vector_type(2))) __bf16 v2bf;
    v2bf r = __builtin_amdgcn_cvt_pk_bf16_f32(a, b);
    return __builtin_bit_cast(unsigned, r);
#else
    typedef __attribute__((ext_vector_type(2))) __bf16 v2bf;
    v2bf r;
    r.x = (__bf16)a;
    r.y = (__bf16)b;
    return __builtin_bit_cast(unsigned, r);
#endif
}
__device__ __forceinline__ unsigned short lo16(unsigned w) { return (unsigned short)w; }
__device__ __forceinline__ unsigned short hi16(unsigned w) { return (unsigned short)(w >> 16); }

// LDS byte offset of a __shared__ object (flat shared pointers keep the LDS
// offset in addr[31:0] per CDNA5 aperture mapping)
__device__ __forceinline__ unsigned lds_off(const void* p) {
    return (unsigned)(size_t)p;
}

__device__ __forceinline__ void async_ld_b128(unsigned lds_dst, unsigned voff,
                                              unsigned long long sbase) {
    asm volatile("global_load_async_to_lds_b128 %0, %1, %2"
                 :: "v"(lds_dst), "v"(voff), "s"(sbase) : "memory");
}
__device__ __forceinline__ void wait_async0() {
#if __has_builtin(__builtin_amdgcn_s_wait_asynccnt)
    __builtin_amdgcn_s_wait_asynccnt(0);
#else
    asm volatile("s_wait_asynccnt 0x0" ::: "memory");
#endif
}

union AFrag { v16bf v; u32x4 q[2]; };

// =====================================================================
// Kernel 1: fused QKV projection.  grid = (M/128, N/64, 3)
//   Y = X @ W + b ; writes bf16 [B,H,T,DK] to ws, fp32 K/V into cache.
// =====================================================================
__global__ __launch_bounds__(256)
void qkv_proj_kernel(const float* __restrict__ Xq, const float* __restrict__ Xk,
                     const float* __restrict__ Xv,
                     const float* __restrict__ Wq, const float* __restrict__ Wk,
                     const float* __restrict__ Wv,
                     const float* __restrict__ bq, const float* __restrict__ bk,
                     const float* __restrict__ bv,
                     unsigned short* __restrict__ qws, unsigned short* __restrict__ kws,
                     unsigned short* __restrict__ vws, float* __restrict__ cache)
{
    const int mode = blockIdx.z;                       // 0=Q 1=K 2=V
    const float* X    = (mode == 0) ? Xq : (mode == 1) ? Xk : Xv;
    const float* W    = (mode == 0) ? Wq : (mode == 1) ? Wk : Wv;
    const float* bias = (mode == 0) ? bq : (mode == 1) ? bk : bv;

    __shared__ __align__(16) unsigned short As[128 * 40];   // 128 rows x 32 k (pad->40)
    __shared__ __align__(16) unsigned short Bs[64 * 40];    // 64 n-rows x 32 k (transposed W)

    const int tid  = threadIdx.x;
    const int lane = tid & 31, wid = tid >> 5;
    const int lh = lane >> 4, ll = lane & 15;
    const int wm = (wid & 3) * 32;                     // wave tile origin in M
    const int wn = (wid >> 2) * 32;                    // wave tile origin in N
    const int m0 = blockIdx.x * 128;
    const int n0 = blockIdx.y * 64;

    v8f acc[2][2] = {};

    for (int kb = 0; kb < DD; kb += 32) {
        __syncthreads();
        // ---- stage A (fp32 -> bf16), 16 elems / thread ----
        {
            const int row = tid >> 1;
            const int c0  = (tid & 1) * 16;
            const float* src = X + (size_t)(m0 + row) * DD + kb + c0;
            union { unsigned w[8]; u32x4 q[2]; } t;
#pragma unroll
            for (int i = 0; i < 4; i++) {
                f32x4 f = *(const f32x4*)(src + i * 4);
                t.w[i * 2 + 0] = f2bf2(f.x, f.y);
                t.w[i * 2 + 1] = f2bf2(f.z, f.w);
            }
            *(u32x4*)&As[row * 40 + c0]     = t.q[0];
            *(u32x4*)&As[row * 40 + c0 + 8] = t.q[1];
        }
        // ---- stage W transposed (fp32 -> bf16), 8 elems / thread ----
        {
            const int kk  = tid >> 3;
            const int nn0 = (tid & 7) * 8;
            const float* src = W + (size_t)(kb + kk) * DD + n0 + nn0;
            f32x4 f0 = *(const f32x4*)src;
            f32x4 f1 = *(const f32x4*)(src + 4);
            const unsigned w0 = f2bf2(f0.x, f0.y), w1 = f2bf2(f0.z, f0.w);
            const unsigned w2 = f2bf2(f1.x, f1.y), w3 = f2bf2(f1.z, f1.w);
            Bs[(nn0 + 0) * 40 + kk] = lo16(w0);
            Bs[(nn0 + 1) * 40 + kk] = hi16(w0);
            Bs[(nn0 + 2) * 40 + kk] = lo16(w1);
            Bs[(nn0 + 3) * 40 + kk] = hi16(w1);
            Bs[(nn0 + 4) * 40 + kk] = lo16(w2);
            Bs[(nn0 + 5) * 40 + kk] = hi16(w2);
            Bs[(nn0 + 6) * 40 + kk] = lo16(w3);
            Bs[(nn0 + 7) * 40 + kk] = hi16(w3);
        }
        __syncthreads();

        AFrag a[2], bf[2];
#pragma unroll
        for (int mi = 0; mi < 2; mi++) {
            const unsigned short* p = &As[(wm + mi * 16 + ll) * 40];
            a[mi].q[0] = *(const u32x4*)(p + lh * 8);
            a[mi].q[1] = *(const u32x4*)(p + 16 + lh * 8);
        }
#pragma unroll
        for (int ni = 0; ni < 2; ni++) {
            const unsigned short* p = &Bs[(wn + ni * 16 + ll) * 40];
            bf[ni].q[0] = *(const u32x4*)(p + lh * 8);
            bf[ni].q[1] = *(const u32x4*)(p + 16 + lh * 8);
        }
#pragma unroll
        for (int mi = 0; mi < 2; mi++)
#pragma unroll
            for (int ni = 0; ni < 2; ni++)
                acc[mi][ni] = __builtin_amdgcn_wmma_f32_16x16x32_bf16(
                    false, a[mi].v, false, bf[ni].v, (short)0, acc[mi][ni], false, false);
    }

    // ---- epilogue ----
#pragma unroll
    for (int mi = 0; mi < 2; mi++)
#pragma unroll
        for (int ni = 0; ni < 2; ni++) {
            const int n  = n0 + wn + ni * 16 + ll;
            const float bb_ = bias[n];
            const int hh = n >> 6, dk = n & 63;
#pragma unroll
            for (int r = 0; r < 8; r++) {
                const int m  = m0 + wm + mi * 16 + r + lh * 8;
                const int bb = m >> 11;            // / T
                const int tt = m & (TT - 1);
                const float val = acc[mi][ni][r] + bb_;
                const size_t qi = (((size_t)(bb * HH + hh) * TT) + tt) * DK + dk;
                if (mode == 0) {
                    qws[qi] = f2bf(val);
                } else if (mode == 1) {
                    kws[qi] = f2bf(val);
                    __builtin_nontemporal_store(
                        val, &cache[(((size_t)(bb * HH + hh) * TT) + tt) * (2 * DK) + dk]);
                } else {
                    vws[qi] = f2bf(val);
                    __builtin_nontemporal_store(
                        val, &cache[(((size_t)(bb * HH + hh) * TT) + tt) * (2 * DK) + DK + dk]);
                }
            }
        }
}

// =====================================================================
// Kernel 2: flash attention.  grid = (T/128, H, B), 8 waves, 16 q-rows/wave
// =====================================================================
__global__ __launch_bounds__(256)
void attn_kernel(const unsigned short* __restrict__ qws, const unsigned short* __restrict__ kws,
                 const unsigned short* __restrict__ vws, const unsigned char* __restrict__ mask,
                 unsigned short* __restrict__ xatt)
{
    __shared__ __align__(16) unsigned short Ks[32 * 72];    // [key][dk] pad 64->72
    __shared__ __align__(16) unsigned short Vt[64 * 40];    // [dk][key] pad 32->40
    __shared__ __align__(16) unsigned short Pl[8 * 16 * 32];// per-wave P tile

    const int tid = threadIdx.x, lane = tid & 31, wid = tid >> 5;
    const int lh = lane >> 4, ll = lane & 15;
    const int h = blockIdx.y, b = blockIdx.z;
    const size_t bh = (size_t)(b * HH + h) * TT;
    const int q0 = blockIdx.x * 128 + wid * 16;

    // Q fragments (16 q-rows x 64 dk) held in registers for the whole loop
    AFrag qa[2];
    {
        const unsigned short* qp = qws + (bh + q0 + ll) * DK;
        qa[0].q[0] = *(const u32x4*)(qp + lh * 8);
        qa[0].q[1] = *(const u32x4*)(qp + 16 + lh * 8);
        qa[1].q[0] = *(const u32x4*)(qp + 32 + lh * 8);
        qa[1].q[1] = *(const u32x4*)(qp + 48 + lh * 8);
    }

    v8f o[4] = {};
    float rmax[8], rsum[8];
#pragma unroll
    for (int r = 0; r < 8; r++) { rmax[r] = -1e30f; rsum[r] = 0.f; }
    const float scale = 0.125f;                 // 1/sqrt(64)
    unsigned short* Pw = &Pl[wid * 512];

    const unsigned long long kbase = (unsigned long long)(size_t)(kws + bh * DK);
    const int kk = tid >> 3, c0 = (tid & 7) * 8;
    const unsigned ks_dst = lds_off(&Ks[kk * 72 + c0]);

    for (int kb = 0; kb < TT; kb += 32) {
        __syncthreads();
        // ---- stage K block [32 keys][64 dk] via async DMA to LDS ----
        async_ld_b128(ks_dst, (unsigned)(((kb + kk) * DK + c0) * 2), kbase);
        // ---- stage V block transposed [64 dk][32 keys] ----
        {
            const unsigned short* src = vws + (bh + kb + kk) * DK + c0;
            union { u32x4 q; unsigned short s[8]; } u;
            u.q = *(const u32x4*)src;
#pragma unroll
            for (int i = 0; i < 8; i++) Vt[(c0 + i) * 40 + kk] = u.s[i];
        }
        if (kb + 32 < TT)
            __builtin_prefetch(vws + (bh + kb + 32 + kk) * DK, 0, 1);
        wait_async0();
        __syncthreads();

        // ---- S = Q K^T : two 16x16 tiles (keys 0-15, 16-31) ----
        v8f s0 = {}, s1 = {};
        {
            const unsigned short* p = &Ks[ll * 72];
            AFrag b0, b1;
            b0.q[0] = *(const u32x4*)(p + lh * 8);
            b0.q[1] = *(const u32x4*)(p + 16 + lh * 8);
            b1.q[0] = *(const u32x4*)(p + 32 + lh * 8);
            b1.q[1] = *(const u32x4*)(p + 48 + lh * 8);
            s0 = __builtin_amdgcn_wmma_f32_16x16x32_bf16(false, qa[0].v, false, b0.v, (short)0, s0, false, false);
            s0 = __builtin_amdgcn_wmma_f32_16x16x32_bf16(false, qa[1].v, false, b1.v, (short)0, s0, false, false);
        }
        {
            const unsigned short* p = &Ks[(16 + ll) * 72];
            AFrag b0, b1;
            b0.q[0] = *(const u32x4*)(p + lh * 8);
            b0.q[1] = *(const u32x4*)(p + 16 + lh * 8);
            b1.q[0] = *(const u32x4*)(p + 32 + lh * 8);
            b1.q[1] = *(const u32x4*)(p + 48 + lh * 8);
            s1 = __builtin_amdgcn_wmma_f32_16x16x32_bf16(false, qa[0].v, false, b0.v, (short)0, s1, false, false);
            s1 = __builtin_amdgcn_wmma_f32_16x16x32_bf16(false, qa[1].v, false, b1.v, (short)0, s1, false, false);
        }

        // ---- online softmax (row = r + 8*lh lives entirely in one half-wave) ----
        const float mb0 = mask[b * TT + kb + ll]      ? 0.f : -1e30f;
        const float mb1 = mask[b * TT + kb + 16 + ll] ? 0.f : -1e30f;
        float al[8];
#pragma unroll
        for (int r = 0; r < 8; r++) {
            float a0 = s0[r] * scale + mb0;
            float a1 = s1[r] * scale + mb1;
            float cm = fmaxf(a0, a1);
#pragma unroll
            for (int off = 8; off > 0; off >>= 1)
                cm = fmaxf(cm, __shfl_xor(cm, off, 16));
            const float nm = fmaxf(rmax[r], cm);
            al[r] = __expf(rmax[r] - nm);
            rmax[r] = nm;
            const float p0 = __expf(a0 - nm);
            const float p1 = __expf(a1 - nm);
            float ps = p0 + p1;
#pragma unroll
            for (int off = 8; off > 0; off >>= 1)
                ps += __shfl_xor(ps, off, 16);
            rsum[r] = rsum[r] * al[r] + ps;
            const int row = r + lh * 8;
            const unsigned w = f2bf2(p0, p1);
            Pw[row * 32 + ll]      = lo16(w);
            Pw[row * 32 + 16 + ll] = hi16(w);
        }
#pragma unroll
        for (int t = 0; t < 4; t++)
#pragma unroll
            for (int r = 0; r < 8; r++) o[t][r] *= al[r];

        asm volatile("s_wait_dscnt 0" ::: "memory");   // P LDS write -> read, same wave

        // ---- O += P V ----
        AFrag pf;
        {
            const unsigned short* p = &Pw[ll * 32];
            pf.q[0] = *(const u32x4*)(p + lh * 8);
            pf.q[1] = *(const u32x4*)(p + 16 + lh * 8);
        }
#pragma unroll
        for (int t = 0; t < 4; t++) {
            AFrag vf;
            const unsigned short* p = &Vt[(t * 16 + ll) * 40];
            vf.q[0] = *(const u32x4*)(p + lh * 8);
            vf.q[1] = *(const u32x4*)(p + 16 + lh * 8);
            o[t] = __builtin_amdgcn_wmma_f32_16x16x32_bf16(false, pf.v, false, vf.v, (short)0, o[t], false, false);
        }
    }

    // ---- normalize + write bf16 [B,T,D] ----
#pragma unroll
    for (int r = 0; r < 8; r++) {
        const float inv = 1.0f / rsum[r];
        const int m = q0 + r + lh * 8;
        const size_t base = ((size_t)b * TT + m) * DD + h * DK;
#pragma unroll
        for (int t = 0; t < 2; t++) {
            const unsigned w = f2bf2(o[2 * t][r] * inv, o[2 * t + 1][r] * inv);
            xatt[base + (2 * t) * 16 + ll]     = lo16(w);
            xatt[base + (2 * t + 1) * 16 + ll] = hi16(w);
        }
    }
}

// =====================================================================
// Kernel 3: output projection (bf16 X @ fp32 Wo + bo -> fp32 out)
// =====================================================================
__global__ __launch_bounds__(256)
void out_proj_kernel(const unsigned short* __restrict__ X, const float* __restrict__ W,
                     const float* __restrict__ bias, float* __restrict__ out)
{
    __shared__ __align__(16) unsigned short As[128 * 40];
    __shared__ __align__(16) unsigned short Bs[64 * 40];

    const int tid = threadIdx.x, lane = tid & 31, wid = tid >> 5;
    const int lh = lane >> 4, ll = lane & 15;
    const int wm = (wid & 3) * 32, wn = (wid >> 2) * 32;
    const int m0 = blockIdx.x * 128, n0 = blockIdx.y * 64;

    v8f acc[2][2] = {};

    const int arow = tid >> 1;
    const int ac0  = (tid & 1) * 16;
    const unsigned as_dst = lds_off(&As[arow * 40 + ac0]);
    const unsigned long long xbase = (unsigned long long)(size_t)X;

    for (int kb = 0; kb < DD; kb += 32) {
        __syncthreads();
        // ---- stage A (bf16 copy) via async DMA to LDS: 2 x b128 / thread ----
        {
            const unsigned voff = (unsigned)((((m0 + arow) * DD) + kb + ac0) * 2);
            async_ld_b128(as_dst,      voff,      xbase);
            async_ld_b128(as_dst + 16, voff + 16, xbase);
        }
        // ---- stage W transposed (fp32 -> bf16) ----
        {
            const int kk = tid >> 3, nn0 = (tid & 7) * 8;
            const float* src = W + (size_t)(kb + kk) * DD + n0 + nn0;
            f32x4 f0 = *(const f32x4*)src;
            f32x4 f1 = *(const f32x4*)(src + 4);
            const unsigned w0 = f2bf2(f0.x, f0.y), w1 = f2bf2(f0.z, f0.w);
            const unsigned w2 = f2bf2(f1.x, f1.y), w3 = f2bf2(f1.z, f1.w);
            Bs[(nn0 + 0) * 40 + kk] = lo16(w0);
            Bs[(nn0 + 1) * 40 + kk] = hi16(w0);
            Bs[(nn0 + 2) * 40 + kk] = lo16(w1);
            Bs[(nn0 + 3) * 40 + kk] = hi16(w1);
            Bs[(nn0 + 4) * 40 + kk] = lo16(w2);
            Bs[(nn0 + 5) * 40 + kk] = hi16(w2);
            Bs[(nn0 + 6) * 40 + kk] = lo16(w3);
            Bs[(nn0 + 7) * 40 + kk] = hi16(w3);
        }
        wait_async0();
        __syncthreads();

        AFrag a[2], bf[2];
#pragma unroll
        for (int mi = 0; mi < 2; mi++) {
            const unsigned short* p = &As[(wm + mi * 16 + ll) * 40];
            a[mi].q[0] = *(const u32x4*)(p + lh * 8);
            a[mi].q[1] = *(const u32x4*)(p + 16 + lh * 8);
        }
#pragma unroll
        for (int ni = 0; ni < 2; ni++) {
            const unsigned short* p = &Bs[(wn + ni * 16 + ll) * 40];
            bf[ni].q[0] = *(const u32x4*)(p + lh * 8);
            bf[ni].q[1] = *(const u32x4*)(p + 16 + lh * 8);
        }
#pragma unroll
        for (int mi = 0; mi < 2; mi++)
#pragma unroll
            for (int ni = 0; ni < 2; ni++)
                acc[mi][ni] = __builtin_amdgcn_wmma_f32_16x16x32_bf16(
                    false, a[mi].v, false, bf[ni].v, (short)0, acc[mi][ni], false, false);
    }

#pragma unroll
    for (int mi = 0; mi < 2; mi++)
#pragma unroll
        for (int ni = 0; ni < 2; ni++) {
            const int n = n0 + wn + ni * 16 + ll;
            const float bb_ = bias[n];
#pragma unroll
            for (int r = 0; r < 8; r++) {
                const int m = m0 + wm + mi * 16 + r + lh * 8;
                __builtin_nontemporal_store(acc[mi][ni][r] + bb_, &out[(size_t)m * DD + n]);
            }
        }
}

// =====================================================================
// launcher
// =====================================================================
extern "C" void kernel_launch(void* const* d_in, const int* in_sizes, int n_in,
                              void* d_out, int out_size, void* d_ws, size_t ws_size,
                              hipStream_t stream) {
    (void)in_sizes; (void)n_in; (void)out_size; (void)ws_size;
    const float* q  = (const float*)d_in[0];
    const float* k  = (const float*)d_in[1];
    const float* v  = (const float*)d_in[2];
    const unsigned char* mask = (const unsigned char*)d_in[3];
    const float* Wq = (const float*)d_in[4];
    const float* bq = (const float*)d_in[5];
    const float* Wk = (const float*)d_in[6];
    const float* bk = (const float*)d_in[7];
    const float* Wv = (const float*)d_in[8];
    const float* bv = (const float*)d_in[9];
    const float* Wo = (const float*)d_in[10];
    const float* bo = (const float*)d_in[11];

    float* out   = (float*)d_out;
    float* cache = out + (size_t)BB * TT * DD;             // new_cache region

    unsigned short* ws   = (unsigned short*)d_ws;
    unsigned short* qws  = ws;                              // bf16 [B,H,T,DK]
    unsigned short* kws  = ws + (size_t)BB * HH * TT * DK;
    unsigned short* vws  = kws + (size_t)BB * HH * TT * DK;
    unsigned short* xatt = vws + (size_t)BB * HH * TT * DK; // bf16 [B,T,D]

    dim3 blk(256);
    qkv_proj_kernel<<<dim3((BB * TT) / 128, DD / 64, 3), blk, 0, stream>>>(
        q, k, v, Wq, Wk, Wv, bq, bk, bv, qws, kws, vws, cache);
    attn_kernel<<<dim3(TT / 128, HH, BB), blk, 0, stream>>>(qws, kws, vws, mask, xatt);
    out_proj_kernel<<<dim3((BB * TT) / 128, DD / 64), blk, 0, stream>>>(xatt, Wo, bo, out);
}
